// SelfAttentionBlock_17171279250114
// MI455X (gfx1250) — matrix operand order repaired and checked by reading, not compiled, not run on hardware
//
#include <hip/hip_runtime.h>

// MI455X (gfx1250) flash-attention self-attention block.
// B=4, H=W=64 -> N=4096, C=256, d=32. bf16 WMMA (16x16x32) with f32 accum.
// Attention kernel: 4 waves/WG share double-buffered K/V LDS tiles staged via
// async global->LDS (ASYNCcnt) to cut L2 traffic 4x and overlap load/compute.

typedef __attribute__((ext_vector_type(16))) __bf16        v16bf;
typedef __attribute__((ext_vector_type(8)))  __bf16        v8bf;
typedef __attribute__((ext_vector_type(8)))  float         v8f;
typedef __attribute__((ext_vector_type(8)))  unsigned short v8us;

#define BN    16384   // B*N flattened rows
#define CIN   256
#define DQK   32
#define NCOLS 320     // 32 (q) + 32 (k) + 256 (v)
#define NKEY  4096

__device__ __forceinline__ unsigned short f2bf_bits(float f) {
  union { float f; unsigned u; } v; v.f = f;
  unsigned r = (v.u + 0x7FFFu + ((v.u >> 16) & 1u)) >> 16;
  return (unsigned short)r;
}

__device__ __forceinline__ v16bf cat8(v8bf lo, v8bf hi) {
  return __builtin_shufflevector(lo, hi, 0,1,2,3,4,5,6,7,8,9,10,11,12,13,14,15);
}

// Low 32 bits of a generic pointer to __shared__ are the LDS byte offset
// (LDS aperture maps addr[31:0] -> LDS). Used as VDST for async-to-LDS.
__device__ __forceinline__ unsigned lds_off32(const void* p) {
  return (unsigned)(unsigned long long)p;
}

// Async 16B global -> LDS copy, tracked by ASYNCcnt (no VGPR round trip).
__device__ __forceinline__ void async_b128(unsigned lds_byte, const void* gptr) {
  unsigned long long ga = (unsigned long long)gptr;
  asm volatile("global_load_async_to_lds_b128 %0, %1, off"
               :: "v"(lds_byte), "v"(ga) : "memory");
}

__device__ __forceinline__ void wait_async0() {
  asm volatile("s_wait_asynccnt 0x0" ::: "memory");
}

// ---------------------------------------------------------------------------
// Kernel 0: x -> bf16; transposed combined weight Wt[320][256] bf16; bias[320]
// ---------------------------------------------------------------------------
__global__ void prep_kernel(const float* __restrict__ x,
                            const float* __restrict__ wq, const float* __restrict__ bq,
                            const float* __restrict__ wk, const float* __restrict__ bk,
                            const float* __restrict__ wv, const float* __restrict__ bv,
                            unsigned short* __restrict__ xb,
                            unsigned short* __restrict__ Wt,
                            float* __restrict__ biasc) {
  long long idx = (long long)blockIdx.x * blockDim.x + threadIdx.x;
  const long long nx = (long long)BN * CIN;
  const long long nw = (long long)NCOLS * CIN;
  if (idx < nx) {
    xb[idx] = f2bf_bits(x[idx]);
  } else if (idx < nx + nw) {
    long long t = idx - nx;
    int col = (int)(t / CIN);
    int k   = (int)(t % CIN);
    float w;
    if (col < 32)      w = wq[k * 32 + col];
    else if (col < 64) w = wk[k * 32 + (col - 32)];
    else               w = wv[k * 256 + (col - 64)];
    Wt[col * CIN + k] = f2bf_bits(w);
  } else if (idx < nx + nw + NCOLS) {
    int i = (int)(idx - nx - nw);
    biasc[i] = (i < 32) ? bq[i] : (i < 64) ? bk[i - 32] : bv[i - 64];
  }
}

// ---------------------------------------------------------------------------
// Kernel 1: QKV projection GEMM. One wave = one 16x16 output tile.
// Q,K stored row-major [BN][32] bf16; V stored transposed Vt[B][256][4096] bf16.
// ---------------------------------------------------------------------------
__global__ void __launch_bounds__(32)
qkv_kernel(const unsigned short* __restrict__ xb,
           const unsigned short* __restrict__ Wt,
           const float* __restrict__ biasc,
           unsigned short* __restrict__ Qb,
           unsigned short* __restrict__ Kb,
           unsigned short* __restrict__ Vt) {
  const int ct   = blockIdx.x;       // 0..19 column tile
  const int m0   = blockIdx.y * 16;  // global flat row base
  const int lane = threadIdx.x;
  const int r    = lane & 15;
  const int half = lane >> 4;

  const int col = ct * 16 + r;       // B-fragment column for this lane
  v8f acc = {};
  const unsigned short* arow = xb + (m0 + r) * CIN;
  const unsigned short* wrow = Wt + col * CIN;
#pragma unroll
  for (int k0 = 0; k0 < CIN; k0 += 32) {
    v16bf A  = *(const v16bf*)(arow + k0 + 16 * half);
    v16bf B  = *(const v16bf*)(wrow + k0 + 16 * half);
    acc = __builtin_amdgcn_wmma_f32_16x16x32_bf16(false, A, false, B,
                                                  (short)0, acc, false, false);
  }
  const float bias = biasc[col];
  if (ct < 4) {
    unsigned short* dst = (ct < 2) ? Qb : Kb;
    const int dcol = ((ct & 1) << 4) + r;
#pragma unroll
    for (int j = 0; j < 8; ++j) {
      int row = m0 + 8 * half + j;
      dst[row * DQK + dcol] = f2bf_bits(acc[j] + bias);
    }
  } else {
    const int c  = col - 64;
    const int b  = m0 >> 12;
    const int nn = (m0 & 4095) + 8 * half;
    v8us pk;
#pragma unroll
    for (int j = 0; j < 8; ++j) pk[j] = f2bf_bits(acc[j] + bias);
    *(v8us*)(Vt + (size_t)b * 256 * NKEY + (size_t)c * NKEY + nn) = pk;
  }
}

// ---------------------------------------------------------------------------
// Kernel 2: flash attention. 4 waves/block, each owning a 16-row Q block.
// K/V 32-key chunks double-buffered in LDS via async global->LDS.
// Per wave per chunk: 2 wmma (S) -> online softmax -> P via LDS -> 16 wmma (PV)
// ---------------------------------------------------------------------------
__global__ void __launch_bounds__(128)
attn_kernel(const unsigned short* __restrict__ Qb,
            const unsigned short* __restrict__ Kb,
            const unsigned short* __restrict__ Vt,
            const float* __restrict__ x,
            float* __restrict__ out) {
  const int tid  = threadIdx.x;
  const int w    = tid >> 5;          // wave id 0..3
  const int lane = tid & 31;
  const int r    = lane & 15;
  const int half = lane >> 4;
  const int m0   = blockIdx.x * 64 + w * 16;  // global flat row base
  const int b    = blockIdx.x >> 6;           // batch (64 blocks per batch)

  __shared__ __align__(128) unsigned short kbuf[2][32 * 32];    // [key][d]
  __shared__ __align__(128) unsigned short vbuf[2][256 * 32];   // [c][key]
  __shared__ __align__(128) unsigned short pt[4][16 * 32];      // per-wave P

  const unsigned short* Kbb = Kb + (size_t)b * NKEY * DQK;
  const unsigned short* Vtb = Vt + (size_t)b * 256 * NKEY;

  // Q A-fragment (d = 32 == WMMA K exactly)
  const unsigned short* qrow = Qb + (m0 + r) * DQK;
  v16bf qa = *(const v16bf*)(qrow + 16 * half);

  v8f acc[16];
#pragma unroll
  for (int i = 0; i < 16; ++i) acc[i] = (v8f){};
  float mrun[8], lrun[8];
#pragma unroll
  for (int j = 0; j < 8; ++j) { mrun[j] = -__builtin_inff(); lrun[j] = 0.f; }

  // Stage one 32-key chunk: K tile 2KB (1 x b128/thread), V tile 16KB (8/thread)
  auto prefetch = [&](int kv0, int buf) {
    {
      const int key = tid >> 2, seg = tid & 3;
      async_b128(lds_off32(&kbuf[buf][key * 32 + seg * 8]),
                 Kbb + (size_t)(kv0 + key) * DQK + seg * 8);
    }
#pragma unroll
    for (int q = 0; q < 2; ++q) {
      const int c = tid * 2 + q;
      const unsigned short* g = Vtb + (size_t)c * NKEY + kv0;
#pragma unroll
      for (int seg = 0; seg < 4; ++seg)
        async_b128(lds_off32(&vbuf[buf][c * 32 + seg * 8]), g + seg * 8);
    }
  };

  prefetch(0, 0);

  for (int it = 0; it < NKEY / 32; ++it) {
    const int buf = it & 1;
    wait_async0();       // this wave's staged loads (chunk `it`) are in LDS
    __syncthreads();     // everyone's loads done; prior buffer fully consumed
    if (it + 1 < NKEY / 32) prefetch((it + 1) * 32, buf ^ 1);

    const unsigned short* kb = kbuf[buf];
    const unsigned short* vb = vbuf[buf];

    // --- S = Q K^T for 32 keys (two 16x16 tiles) ---
    v16bf kb0 = *(const v16bf*)(kb + (r) * 32 + 16 * half);
    v16bf kb1 = *(const v16bf*)(kb + (16 + r) * 32 + 16 * half);
    v8f z = {};
    v8f s0 = __builtin_amdgcn_wmma_f32_16x16x32_bf16(false, qa, false, kb0,
                                                     (short)0, z, false, false);
    v8f s1 = __builtin_amdgcn_wmma_f32_16x16x32_bf16(false, qa, false, kb1,
                                                     (short)0, z, false, false);
    // --- online softmax (row = 8*half + j; reduce across a 16-lane half) ---
    float alpha[8];
#pragma unroll
    for (int j = 0; j < 8; ++j) {
      float mx = fmaxf(s0[j], s1[j]);
      mx = fmaxf(mx, __shfl_xor(mx, 1));
      mx = fmaxf(mx, __shfl_xor(mx, 2));
      mx = fmaxf(mx, __shfl_xor(mx, 4));
      mx = fmaxf(mx, __shfl_xor(mx, 8));
      float mnew = fmaxf(mrun[j], mx);
      float p0 = __expf(s0[j] - mnew);
      float p1 = __expf(s1[j] - mnew);
      float rs = p0 + p1;
      rs += __shfl_xor(rs, 1);
      rs += __shfl_xor(rs, 2);
      rs += __shfl_xor(rs, 4);
      rs += __shfl_xor(rs, 8);
      float a = __expf(mrun[j] - mnew);
      lrun[j] = lrun[j] * a + rs;
      mrun[j] = mnew;
      alpha[j] = a;
      const int row = 8 * half + j;
      pt[w][row * 32 + r]      = f2bf_bits(p0);
      pt[w][row * 32 + 16 + r] = f2bf_bits(p1);
    }
#pragma unroll
    for (int ctile = 0; ctile < 16; ++ctile)
#pragma unroll
      for (int j = 0; j < 8; ++j) acc[ctile][j] *= alpha[j];

    // P back out of LDS in A-fragment layout (16x32, K = 32 keys); same-wave
    // DS ordering makes this safe without a barrier.
    v16bf pa = *(const v16bf*)(&pt[w][r * 32 + 16 * half]);

    // --- O += P V : 16 column tiles of 16 channels, V from LDS ---
#pragma unroll
    for (int ctile = 0; ctile < 16; ++ctile) {
      v16bf vfrag = *(const v16bf*)(vb + (ctile * 16 + r) * 32 + 16 * half);
      acc[ctile] = __builtin_amdgcn_wmma_f32_16x16x32_bf16(false, pa, false, vfrag,
                                                           (short)0, acc[ctile],
                                                           false, false);
    }
  }

  // --- epilogue: normalize, add residual, coalesced stores ---
  float invl[8];
#pragma unroll
  for (int j = 0; j < 8; ++j) invl[j] = 1.0f / lrun[j];
#pragma unroll
  for (int ctile = 0; ctile < 16; ++ctile) {
#pragma unroll
    for (int j = 0; j < 8; ++j) {
      const int row = m0 + 8 * half + j;
      const int c   = ctile * 16 + r;
      const size_t o = (size_t)row * 256 + c;
      out[o] = acc[ctile][j] * invl[j] + x[o];
    }
  }
}

// ---------------------------------------------------------------------------
extern "C" void kernel_launch(void* const* d_in, const int* in_sizes, int n_in,
                              void* d_out, int out_size, void* d_ws, size_t ws_size,
                              hipStream_t stream) {
  const float* x  = (const float*)d_in[0];
  const float* wq = (const float*)d_in[1];
  const float* bq = (const float*)d_in[2];
  const float* wk = (const float*)d_in[3];
  const float* bk = (const float*)d_in[4];
  const float* wv = (const float*)d_in[5];
  const float* bv = (const float*)d_in[6];
  float* out = (float*)d_out;

  // workspace layout (bytes, all 256B-aligned)
  char* ws = (char*)d_ws;
  unsigned short* xb    = (unsigned short*)(ws + 0);                 // 8,388,608
  unsigned short* Wt    = (unsigned short*)(ws + 8388608);           //   163,840
  float*          biasc = (float*)        (ws + 8552448);           //     1,280
  unsigned short* Qb    = (unsigned short*)(ws + 8553728);           // 1,048,576
  unsigned short* Kb    = (unsigned short*)(ws + 9602304);           // 1,048,576
  unsigned short* Vt    = (unsigned short*)(ws + 10650880);          // 8,388,608

  {
    long long total = (long long)BN * CIN + (long long)NCOLS * CIN + NCOLS;
    int blocks = (int)((total + 255) / 256);
    prep_kernel<<<blocks, 256, 0, stream>>>(x, wq, bq, wk, bk, wv, bv,
                                            xb, Wt, biasc);
  }
  {
    dim3 grid(20, 1024);
    qkv_kernel<<<grid, 32, 0, stream>>>(xb, Wt, biasc, Qb, Kb, Vt);
  }
  {
    // 256 blocks x 128 threads: 4 Q blocks (64 rows) share K/V LDS stream
    attn_kernel<<<256, 128, 0, stream>>>(Qb, Kb, Vt, x, out);
  }
}
